// CosineMultiheadAttention_1726576854999
// MI455X (gfx1250) — matrix operand order repaired and checked by reading, not compiled, not run on hardware
//
#include <hip/hip_runtime.h>

// Cosine multi-head attention for MI455X (gfx1250), wave32 + WMMA f16 +
// async global->LDS staging (ASYNCcnt path).
// S=2048 B=2 E=1024 H=16 HD=64. Compute-bound -> v_wmma_f32_16x16x32_f16.

#define S_LEN 2048
#define BATCH 2
#define EMB   1024
#define NHEAD 16
#define HDIM  64
#define M_TOT (S_LEN * BATCH)   // 4096 token rows
#define TAU_MIN 0.01f
#define NRM_EPS 1e-12f

typedef __attribute__((ext_vector_type(16))) _Float16 v16h;
typedef __attribute__((ext_vector_type(8)))  float    v8f;
typedef __attribute__((ext_vector_type(4)))  float    v4f;

union FragU { v16h v; v4f f[2]; };

__device__ __forceinline__ v8f wmma_f16(v16h a, v16h b, v8f c) {
  // (neg_a, A, neg_b, B, c_mod, C, reuse_a, reuse_b)
  return __builtin_amdgcn_wmma_f32_16x16x32_f16(false, a, false, b, (short)0, c,
                                                false, false);
}

// Async copy of 16 bytes global -> LDS, tracked by ASYNCcnt (ISA 10.7).
// Generic LDS pointers carry the LDS byte offset in their low 32 bits.
__device__ __forceinline__ void async_b128(const _Float16* g, _Float16* l) {
  unsigned lds_off = (unsigned)(uintptr_t)l;
  unsigned long long ga = (unsigned long long)(uintptr_t)g;
  asm volatile("global_load_async_to_lds_b128 %0, %1, off"
               :: "v"(lds_off), "v"(ga) : "memory");
}
__device__ __forceinline__ void async_wait0() {
  asm volatile("s_wait_asynccnt 0" ::: "memory");
}

// A-matrix fragment (MxK, 16x32 f16). Per ISA 7.12.2:
// lanes 0-15 : M=row0+lane,    halves = K[0..7], K[16..23]
// lanes 16-31: M=row0+lane-16, halves = K[8..15], K[24..31]
__device__ __forceinline__ v16h fragA(const _Float16* base, int row0, int stride,
                                      int lane) {
  const int r = row0 + (lane & 15);
  const _Float16* p = base + r * stride + ((lane >> 4) << 3);
  FragU u;
  u.f[0] = *(const v4f*)(p);
  u.f[1] = *(const v4f*)(p + 16);
  return u.v;
}

// B-matrix fragment (KxN, 32x16 f16). Per ISA 7.12.2 (B striped across lanes):
// lanes 0-15 : N=col0+lane,    halves = K[0..7], K[16..23]
// lanes 16-31: N=col0+lane-16, halves = K[8..15], K[24..31]
// `base` points at a tile laid out [N][K] (K contiguous per row).
__device__ __forceinline__ v16h fragB(const _Float16* base, int col0, int stride,
                                      int lane) {
  const int c = col0 + (lane & 15);
  const _Float16* p = base + c * stride + ((lane >> 4) << 3);
  FragU u;
  u.f[0] = *(const v4f*)(p);
  u.f[1] = *(const v4f*)(p + 16);
  return u.v;
}

__device__ __forceinline__ void cvt8(_Float16* d, v4f a, v4f b) {
  d[0] = (_Float16)a.x; d[1] = (_Float16)a.y;
  d[2] = (_Float16)a.z; d[3] = (_Float16)a.w;
  d[4] = (_Float16)b.x; d[5] = (_Float16)b.y;
  d[6] = (_Float16)b.z; d[7] = (_Float16)b.w;
}

// ---------------------------------------------------------------------------
// Kernel 1: packed in-projection GEMM + bias + per-head L2 normalize.
// Grid: (M_TOT/64, 3*EMB/64). K-chunk 64 -> 4 WMMAs/wave between barriers.
// ---------------------------------------------------------------------------
__global__ __launch_bounds__(256) void proj_kernel(
    const float* __restrict__ query, const float* __restrict__ keyx,
    const float* __restrict__ value, const float* __restrict__ W,
    const float* __restrict__ bias, _Float16* __restrict__ qn,
    _Float16* __restrict__ kn, _Float16* __restrict__ vhd) {
  const int m0  = blockIdx.x * 64;
  const int nt  = blockIdx.y;          // 0..47 over 3072 cols
  const int n0  = nt * 64;
  const int mat = nt >> 4;             // 0=q 1=k 2=v
  const int h   = nt & 15;             // head (HD==64 => one tile == one head)
  const float* X = (mat == 0) ? query : (mat == 1 ? keyx : value);

  __shared__ _Float16 ldsA[64 * 72];
  __shared__ _Float16 ldsB[64 * 72];
  __shared__ float    outT[64 * 68];
  __shared__ float    part[64 * 4];
  __shared__ float    rscale[64];

  const int tid  = threadIdx.x;
  const int lane = tid & 31;
  const int wave = tid >> 5;
  const int wr   = (wave & 3) * 16;    // row strip in tile
  const int wc   = (wave >> 2) * 32;   // col strip in tile

  v8f acc0 = {}, acc1 = {};

  const int lrow = tid >> 2;           // 0..63
  const int lk   = (tid & 3) * 16;     // 0,16,32,48

#pragma unroll 1
  for (int kb = 0; kb < EMB; kb += 64) {
    const float* ap = X + (size_t)(m0 + lrow) * EMB + kb + lk;
    const float* bp = W + (size_t)(n0 + lrow) * EMB + kb + lk;
    _Float16* dA = ldsA + lrow * 72 + lk;
    _Float16* dB = ldsB + lrow * 72 + lk;
    cvt8(dA,     ((const v4f*)ap)[0], ((const v4f*)ap)[1]);
    cvt8(dA + 8, ((const v4f*)ap)[2], ((const v4f*)ap)[3]);
    cvt8(dB,     ((const v4f*)bp)[0], ((const v4f*)bp)[1]);
    cvt8(dB + 8, ((const v4f*)bp)[2], ((const v4f*)bp)[3]);
    __syncthreads();
    v16h a0 = fragA(ldsA,      wr, 72, lane);
    v16h a1 = fragA(ldsA + 32, wr, 72, lane);
    acc0 = wmma_f16(a0, fragB(ldsB,      wc,      72, lane), acc0);
    acc0 = wmma_f16(a1, fragB(ldsB + 32, wc,      72, lane), acc0);
    acc1 = wmma_f16(a0, fragB(ldsB,      wc + 16, 72, lane), acc1);
    acc1 = wmma_f16(a1, fragB(ldsB + 32, wc + 16, 72, lane), acc1);
    __syncthreads();
  }

  // bias (per-column => per-lane constant within a 16x16 tile)
  const int pc = lane & 15;
  const int pr = (lane >> 4) * 8;
  const float bv0 = bias[n0 + wc + pc];
  const float bv1 = bias[n0 + wc + 16 + pc];
#pragma unroll
  for (int r = 0; r < 8; ++r) {
    outT[(wr + pr + r) * 68 + wc + pc]      = acc0[r] + bv0;
    outT[(wr + pr + r) * 68 + wc + 16 + pc] = acc1[r] + bv1;
  }
  __syncthreads();

  // per-row L2 norm over the full 64-wide head (q/k only)
  {
    const int row = tid >> 2, quad = tid & 3;
    const float* rp = outT + row * 68 + quad * 16;
    float s = 0.f;
#pragma unroll
    for (int i = 0; i < 16; ++i) { float v = rp[i]; s += v * v; }
    part[row * 4 + quad] = s;
  }
  __syncthreads();
  if (tid < 64) {
    float s = part[tid * 4] + part[tid * 4 + 1] + part[tid * 4 + 2] + part[tid * 4 + 3];
    rscale[tid] = (mat < 2) ? (1.0f / fmaxf(sqrtf(s), NRM_EPS)) : 1.0f;
  }
  __syncthreads();

  // write f16 to [b][h][s][d]
  {
    const int row = tid >> 2, c0 = (tid & 3) * 16;
    const int m = m0 + row;
    const int bidx = m % BATCH, sidx = m / BATCH;
    _Float16* dst = (mat == 0) ? qn : (mat == 1 ? kn : vhd);
    const size_t base = (((size_t)bidx * NHEAD + h) * S_LEN + sidx) * HDIM;
    const float sc = rscale[row];
#pragma unroll
    for (int i = 0; i < 16; ++i)
      dst[base + c0 + i] = (_Float16)(outT[row * 68 + c0 + i] * sc);
  }
}

// ---------------------------------------------------------------------------
// Kernel 2: flash attention per (b,h, 64-query block). 128 threads = 4 waves.
// Q/K tiles staged with global_load_async_to_lds_b128 (ASYNCcnt), V staged
// with a manual transpose. Online softmax matches WMMA C/D layout.
// ---------------------------------------------------------------------------
__global__ __launch_bounds__(128) void attn_kernel(
    const _Float16* __restrict__ qn, const _Float16* __restrict__ kn,
    const _Float16* __restrict__ vhd, const float* __restrict__ tau,
    _Float16* __restrict__ oh) {
  const int q0 = blockIdx.x * 64;
  const int bh = blockIdx.y;                 // b*NHEAD + h
  const int b  = bh >> 4, h = bh & 15;
  const float inv_tau = 1.0f / fmaxf(tau[h], TAU_MIN);

  __shared__ _Float16 Qs[64 * 72];
  __shared__ _Float16 Ks[64 * 72];
  __shared__ _Float16 Vt[64 * 72];           // transposed: [dim][key]
  __shared__ _Float16 Ps[4][16 * 72];        // wave-private P tiles

  const int tid  = threadIdx.x;
  const int lane = tid & 31;
  const int wave = tid >> 5;
  const size_t headbase = (size_t)bh * S_LEN * HDIM;

  const int srow = tid >> 1;                 // staging row 0..63
  const int scol = (tid & 1) * 32;           // staging col 0 / 32 halves

  // stage Q block asynchronously (4 x b128 per thread)
  {
    const _Float16* g = qn + headbase + (size_t)(q0 + srow) * HDIM + scol;
    _Float16* l = Qs + srow * 72 + scol;
#pragma unroll
    for (int i = 0; i < 4; ++i) async_b128(g + 8 * i, l + 8 * i);
  }

  v8f oacc[4] = {};
  v8f sacc[4];
  float m_r[8], l_r[8];
#pragma unroll
  for (int r = 0; r < 8; ++r) { m_r[r] = -1e30f; l_r[r] = 0.f; }

  const int r0 = wave * 16;
  const int pc = lane & 15;
  const int pr = (lane >> 4) * 8;
  _Float16* pw = &Ps[wave][0];

#pragma unroll 1
  for (int j = 0; j < S_LEN; j += 64) {
    __syncthreads();   // previous iteration's LDS reads done
    {  // stage K block row-major [key][dim], async
      const _Float16* g = kn + headbase + (size_t)(j + srow) * HDIM + scol;
      _Float16* l = Ks + srow * 72 + scol;
#pragma unroll
      for (int i = 0; i < 4; ++i) async_b128(g + 8 * i, l + 8 * i);
    }
    {  // stage V transposed [dim][key] (transpose can't use async copy)
      const v4f* s = (const v4f*)(vhd + headbase + (size_t)(j + srow) * HDIM + scol);
      _Float16 tmp[32];
      ((v4f*)tmp)[0] = s[0]; ((v4f*)tmp)[1] = s[1];
      ((v4f*)tmp)[2] = s[2]; ((v4f*)tmp)[3] = s[3];
#pragma unroll
      for (int i = 0; i < 32; ++i) Vt[(scol + i) * 72 + srow] = tmp[i];
    }
    async_wait0();     // Q (first iter) + K tile committed to LDS
    __syncthreads();

    // S = Q @ K^T  (16 rows x 64 keys per wave)
    v16h aq0 = fragA(Qs, r0, 72, lane);
    v16h aq1 = fragA(Qs + 32, r0, 72, lane);
#pragma unroll
    for (int i = 0; i < 4; ++i) {
      v8f s = {};
      s = wmma_f16(aq0, fragB(Ks, i * 16, 72, lane), s);
      s = wmma_f16(aq1, fragB(Ks + 32, i * 16, 72, lane), s);
      sacc[i] = s * inv_tau;
    }

    // online softmax stats (row r lives in lanes 0-15, row r+8 in lanes 16-31)
    float cf[8];
#pragma unroll
    for (int r = 0; r < 8; ++r) {
      float mx = fmaxf(fmaxf(sacc[0][r], sacc[1][r]),
                       fmaxf(sacc[2][r], sacc[3][r]));
      mx = fmaxf(mx, __shfl_xor(mx, 8, 32));
      mx = fmaxf(mx, __shfl_xor(mx, 4, 32));
      mx = fmaxf(mx, __shfl_xor(mx, 2, 32));
      mx = fmaxf(mx, __shfl_xor(mx, 1, 32));
      const float mn = fmaxf(m_r[r], mx);
      cf[r] = __expf(m_r[r] - mn);
      m_r[r] = mn;
    }
#pragma unroll
    for (int i = 0; i < 4; ++i)
#pragma unroll
      for (int r = 0; r < 8; ++r) sacc[i][r] = __expf(sacc[i][r] - m_r[r]);
#pragma unroll
    for (int r = 0; r < 8; ++r) {
      float t = sacc[0][r] + sacc[1][r] + sacc[2][r] + sacc[3][r];
      t += __shfl_xor(t, 8, 32);
      t += __shfl_xor(t, 4, 32);
      t += __shfl_xor(t, 2, 32);
      t += __shfl_xor(t, 1, 32);
      l_r[r] = l_r[r] * cf[r] + t;
#pragma unroll
      for (int i = 0; i < 4; ++i) oacc[i][r] *= cf[r];
    }

    // D-layout -> A-layout through wave-private LDS
#pragma unroll
    for (int r = 0; r < 8; ++r) {
      pw[(pr + r) * 72 + pc]      = (_Float16)sacc[0][r];
      pw[(pr + r) * 72 + pc + 16] = (_Float16)sacc[1][r];
      pw[(pr + r) * 72 + pc + 32] = (_Float16)sacc[2][r];
      pw[(pr + r) * 72 + pc + 48] = (_Float16)sacc[3][r];
    }

    // O += P @ V
    v16h pa0 = fragA(pw, 0, 72, lane);
    v16h pa1 = fragA(pw + 32, 0, 72, lane);
#pragma unroll
    for (int i = 0; i < 4; ++i) {
      oacc[i] = wmma_f16(pa0, fragB(Vt, i * 16, 72, lane), oacc[i]);
      oacc[i] = wmma_f16(pa1, fragB(Vt + 32, i * 16, 72, lane), oacc[i]);
    }
  }

  float inv_l[8];
#pragma unroll
  for (int r = 0; r < 8; ++r) inv_l[r] = 1.0f / l_r[r];
#pragma unroll
  for (int i = 0; i < 4; ++i)
#pragma unroll
    for (int r = 0; r < 8; ++r) {
      const int st = q0 + r0 + pr + r;
      oh[((size_t)st * BATCH + b) * EMB + h * HDIM + i * 16 + pc] =
          (_Float16)(oacc[i][r] * inv_l[r]);
    }
}

// ---------------------------------------------------------------------------
// Kernel 3: out-projection GEMM + bias, fp32 output. A-tile (f16) staged
// asynchronously; weights converted f32->f16 in registers. K-chunk 64.
// ---------------------------------------------------------------------------
__global__ __launch_bounds__(256) void outproj_kernel(
    const _Float16* __restrict__ oh, const float* __restrict__ W,
    const float* __restrict__ bias, float* __restrict__ out) {
  const int m0 = blockIdx.x * 64;
  const int n0 = blockIdx.y * 64;

  __shared__ _Float16 ldsA[64 * 72];
  __shared__ _Float16 ldsB[64 * 72];

  const int tid  = threadIdx.x;
  const int lane = tid & 31;
  const int wave = tid >> 5;
  const int wr   = (wave & 3) * 16;
  const int wc   = (wave >> 2) * 32;

  v8f acc0 = {}, acc1 = {};
  const int lrow = tid >> 2;
  const int lk   = (tid & 3) * 16;

#pragma unroll 1
  for (int kb = 0; kb < EMB; kb += 64) {
    {  // A tile: f16 already, async global->LDS (2 x b128)
      const _Float16* g = oh + (size_t)(m0 + lrow) * EMB + kb + lk;
      _Float16* l = ldsA + lrow * 72 + lk;
      async_b128(g, l);
      async_b128(g + 8, l + 8);
    }
    {  // B tile: convert f32 weights -> f16
      const float* bp = W + (size_t)(n0 + lrow) * EMB + kb + lk;
      _Float16* dB = ldsB + lrow * 72 + lk;
      cvt8(dB,     ((const v4f*)bp)[0], ((const v4f*)bp)[1]);
      cvt8(dB + 8, ((const v4f*)bp)[2], ((const v4f*)bp)[3]);
    }
    async_wait0();
    __syncthreads();
    v16h a0 = fragA(ldsA,      wr, 72, lane);
    v16h a1 = fragA(ldsA + 32, wr, 72, lane);
    acc0 = wmma_f16(a0, fragB(ldsB,      wc,      72, lane), acc0);
    acc0 = wmma_f16(a1, fragB(ldsB + 32, wc,      72, lane), acc0);
    acc1 = wmma_f16(a0, fragB(ldsB,      wc + 16, 72, lane), acc1);
    acc1 = wmma_f16(a1, fragB(ldsB + 32, wc + 16, 72, lane), acc1);
    __syncthreads();
  }

  const int pc = lane & 15;
  const int pr = (lane >> 4) * 8;
  const float bv0 = bias[n0 + wc + pc];
  const float bv1 = bias[n0 + wc + 16 + pc];
#pragma unroll
  for (int r = 0; r < 8; ++r) {
    const size_t row = (size_t)(m0 + wr + pr + r);
    out[row * EMB + n0 + wc + pc]      = acc0[r] + bv0;
    out[row * EMB + n0 + wc + 16 + pc] = acc1[r] + bv1;
  }
}

// ---------------------------------------------------------------------------
extern "C" void kernel_launch(void* const* d_in, const int* in_sizes, int n_in,
                              void* d_out, int out_size, void* d_ws, size_t ws_size,
                              hipStream_t stream) {
  const float* query = (const float*)d_in[0];
  const float* keyx  = (const float*)d_in[1];
  const float* value = (const float*)d_in[2];
  const float* w_in  = (const float*)d_in[3];
  const float* b_in  = (const float*)d_in[4];
  const float* w_out = (const float*)d_in[5];
  const float* b_out = (const float*)d_in[6];
  const float* tau   = (const float*)d_in[7];
  float* out = (float*)d_out;

  const size_t NELEM = (size_t)S_LEN * BATCH * EMB;  // 4,194,304
  _Float16* ws  = (_Float16*)d_ws;
  _Float16* qn  = ws;              // [b][h][s][d] f16, L2-normalized q
  _Float16* kn  = qn + NELEM;      // [b][h][s][d] f16, L2-normalized k
  _Float16* vhd = kn + NELEM;      // [b][h][s][d] f16
  _Float16* oh  = vhd + NELEM;     // [s][b][e]    f16 attention output
  (void)in_sizes; (void)n_in; (void)out_size; (void)ws_size;

  dim3 gp(M_TOT / 64, (3 * EMB) / 64);          // 64 x 48
  proj_kernel<<<gp, 256, 0, stream>>>(query, keyx, value, w_in, b_in, qn, kn, vhd);

  dim3 ga(S_LEN / 64, BATCH * NHEAD);           // 32 x 32
  attn_kernel<<<ga, 128, 0, stream>>>(qn, kn, vhd, tau, oh);

  dim3 go(M_TOT / 64, EMB / 64);                // 64 x 16
  outproj_kernel<<<go, 256, 0, stream>>>(oh, w_out, b_out, out);
}
